// RelativeAttention_28561532518975
// MI455X (gfx1250) — matrix operand order repaired and checked by reading, not compiled, not run on hardware
//
#include <hip/hip_runtime.h>

#define L_SEQ   2048
#define D_MODEL 512
#define N_HEADS 8
#define D_HEAD  64
#define N_BATCH 2
#define LSTRIDE 2052   // 16-row logits stride in LDS (bank-skewed, mult of 4)

typedef __attribute__((ext_vector_type(16))) __bf16 v16bf;
typedef __attribute__((ext_vector_type(8)))  __bf16 v8bf;
typedef __attribute__((ext_vector_type(4)))  __bf16 v4bf;
typedef __attribute__((ext_vector_type(8)))  float  v8f;
typedef __attribute__((ext_vector_type(4)))  float  v4f;

__device__ __forceinline__ v8f wmma_bf16(v16bf a, v16bf b, v8f c) {
  // D = A(16x32 bf16) * B(32x16 bf16) + C(16x16 f32)
  return __builtin_amdgcn_wmma_f32_16x16x32_bf16(false, a, false, b, (short)0, c,
                                                 false, false);
}

// Assemble a 16-bit A-fragment from its two contiguous 8-element chunks
// (contiguous order: lane<16 holds K {0..15}, lane>=16 holds K {16..31})
__device__ __forceinline__ v16bf a_frag(const __bf16* row, int k0, int hi) {
  v8bf lo = *(const v8bf*)(row + k0 + (hi << 4));
  v8bf hh = *(const v8bf*)(row + k0 + 8 + (hi << 4));
  return __builtin_shufflevector(lo, hh, 0, 1, 2, 3, 4, 5, 6, 7,
                                 8, 9, 10, 11, 12, 13, 14, 15);
}

// ---------------------------------------------------------- f32 -> bf16 (x4)
__global__ void cvt_kernel(const float* __restrict__ X, __bf16* __restrict__ Y,
                           int n4) {
  int idx = blockIdx.x * blockDim.x + threadIdx.x;
  if (idx < n4) {
    v4f v = ((const v4f*)X)[idx];
    v4bf o;
    o[0] = (__bf16)v[0]; o[1] = (__bf16)v[1]; o[2] = (__bf16)v[2]; o[3] = (__bf16)v[3];
    ((v4bf*)Y)[idx] = o;
  }
}

// ------------------------------- W(512x512 f32) -> Wt[n][k] bf16 (transposed)
__global__ __launch_bounds__(256)
void wtrans_kernel(const float* __restrict__ W, __bf16* __restrict__ Wt) {
  __shared__ float t[32][33];
  const int tx = threadIdx.x & 31, ty = threadIdx.x >> 5;   // 32x8
  const int k0 = blockIdx.x * 32, n0 = blockIdx.y * 32;
#pragma unroll
  for (int s = 0; s < 32; s += 8)
    t[ty + s][tx] = W[(size_t)(k0 + ty + s) * D_MODEL + n0 + tx];
  __syncthreads();
#pragma unroll
  for (int s = 0; s < 32; s += 8)
    Wt[(size_t)(n0 + ty + s) * D_MODEL + k0 + tx] = (__bf16)t[tx][ty + s];
}

// --------------------------- Vh[bh][l][dh] bf16 -> Vt[bh][dh][l] (transposed)
__global__ __launch_bounds__(256)
void vtrans_kernel(const __bf16* __restrict__ Vh, __bf16* __restrict__ Vt) {
  __shared__ __bf16 t[32][33];
  const int tx = threadIdx.x & 31, ty = threadIdx.x >> 5;
  const int bh = blockIdx.z;
  const int l0 = blockIdx.x * 32, d0 = blockIdx.y * 32;
  const __bf16* src = Vh + (size_t)bh * L_SEQ * D_HEAD;
  __bf16* dst = Vt + (size_t)bh * D_HEAD * L_SEQ;
#pragma unroll
  for (int s = 0; s < 32; s += 8)
    t[ty + s][tx] = src[(size_t)(l0 + ty + s) * D_HEAD + d0 + tx];
  __syncthreads();
#pragma unroll
  for (int s = 0; s < 32; s += 8)
    dst[(size_t)(d0 + ty + s) * L_SEQ + l0 + tx] = t[tx][ty + s];
}

// ---------------------------------------------------- projection: Y = X@W + b
// Xb: (B*L, D) bf16, Wt: transposed bf16 [n][k], out: bf16 head layout [b,h,l,dh]
__global__ __launch_bounds__(256)
void proj_kernel(const __bf16* __restrict__ Xb, const __bf16* __restrict__ Wt,
                 const float* __restrict__ bias, __bf16* __restrict__ Yh) {
  const int wid = threadIdx.x >> 5, lane = threadIdx.x & 31;
  const int hi = lane >> 4, l15 = lane & 15;
  const int gw = blockIdx.x * 8 + wid;          // 8192 tiles
  const int mt = gw >> 5, nt = gw & 31;
  const int m0 = mt * 16, n0 = nt * 16;

  v8f acc = {};
  const __bf16* xr = Xb + (size_t)(m0 + l15) * D_MODEL;
  const __bf16* wr = Wt + (size_t)(n0 + l15) * D_MODEL;
  for (int k0 = 0; k0 < D_MODEL; k0 += 32) {
    v16bf a = a_frag(xr, k0, hi);
    v16bf b = *(const v16bf*)(wr + k0 + (hi << 4));
    acc = wmma_bf16(a, b, acc);
  }
  const int n = n0 + l15, h = n >> 6, dh = n & 63;
  const float bv = bias[n];
#pragma unroll
  for (int r = 0; r < 8; ++r) {
    const int m = m0 + r + (hi << 3);
    const int bb = m >> 11, l = m & (L_SEQ - 1);
    Yh[(((size_t)(bb * N_HEADS + h) * L_SEQ) + l) * D_HEAD + dh] =
        (__bf16)(acc[r] + bv);
  }
}

// --------------------- fused QK^T + skewed Srel + softmax + attn@V per stripe
__global__ __launch_bounds__(256)
void attn_kernel(const __bf16* __restrict__ Qh, const __bf16* __restrict__ Kh,
                 const __bf16* __restrict__ Vt, const __bf16* __restrict__ Eb,
                 float* __restrict__ attnW, __bf16* __restrict__ ctx) {
  extern __shared__ float smem[];
  float* logits = smem;                    // 16 * LSTRIDE
  float* tscr   = logits + 16 * LSTRIDE;   // 8 waves * 16*32
  float* red    = tscr + 8 * 512;          // 16*16
  float* rowmax = red + 256;               // 16
  float* rowsum = rowmax + 16;             // 16
  float* cpart  = rowsum + 16;             // 8 waves * 16*64

  const int tid = threadIdx.x;
  const int wid = tid >> 5, lane = tid & 31;
  const int hi = lane >> 4, l15 = lane & 15;
  const int bh = blockIdx.x >> 7;          // batch*head
  const int ti = blockIdx.x & 127;         // 16-row tile

  const __bf16* Qb  = Qh + (size_t)bh * L_SEQ * D_HEAD;
  const __bf16* Kb  = Kh + (size_t)bh * L_SEQ * D_HEAD;
  const __bf16* Vtb = Vt + (size_t)bh * D_HEAD * L_SEQ;

  // Q fragments for this stripe (DH=64 -> two k-steps), reused by QK and QE
  const __bf16* qr = Qb + (size_t)(ti * 16 + l15) * D_HEAD;
  v16bf qa[2];
#pragma unroll
  for (int ks = 0; ks < 2; ++ks) qa[ks] = a_frag(qr, ks * 32, hi);

  float* ts = tscr + wid * 512;
  for (int tj = wid; tj <= ti; tj += 8) {
    if (tj + 8 <= ti)  // prefetch next K tile for this wave
      __builtin_prefetch(Kb + (size_t)((tj + 8) * 16 + l15) * D_HEAD, 0, 3);
    v8f acc = {}, t0 = {}, t1 = {};
    const int base = (L_SEQ - 16) + 16 * (tj - ti);   // >= 0
#pragma unroll
    for (int ks = 0; ks < 2; ++ks) {
      const int kb = ks * 32 + (hi << 4);
      // K^T fragment: B[k][n] = K[row n][k] -> contiguous 16 bf16 per lane
      v16bf bk = *(const v16bf*)(Kb + (size_t)(tj * 16 + l15) * D_HEAD + kb);
      acc = wmma_bf16(qa[ks], bk, acc);
      // E-span fragments for Srel (rows base..base+31, clamped; overflow rows
      // only feed causally-masked positions)
      int er0 = base + l15;      if (er0 > L_SEQ - 1) er0 = L_SEQ - 1;
      int er1 = base + 16 + l15; if (er1 > L_SEQ - 1) er1 = L_SEQ - 1;
      v16bf be0 = *(const v16bf*)(Eb + (size_t)er0 * D_HEAD + kb);
      v16bf be1 = *(const v16bf*)(Eb + (size_t)er1 * D_HEAD + kb);
      t0 = wmma_bf16(qa[ks], be0, t0);
      t1 = wmma_bf16(qa[ks], be1, t1);
    }
    // bounce T(16x32) through LDS to apply the skew: Srel[i,j]=T[i,15-i+j]
#pragma unroll
    for (int r = 0; r < 8; ++r) {
      const int ip = r + (hi << 3);
      ts[ip * 32 + l15]      = t0[r];
      ts[ip * 32 + 16 + l15] = t1[r];
    }
#pragma unroll
    for (int r = 0; r < 8; ++r) {
      const int ip = r + (hi << 3);
      const float srel = ts[ip * 32 + (15 - ip + l15)];
      logits[ip * LSTRIDE + tj * 16 + l15] = (acc[r] + srel) * 0.125f;
    }
  }
  __syncthreads();

  // causal softmax: row i = ti*16+r, valid columns 0..i ; 16 threads per row
  const int r = tid >> 4, c = tid & 15;
  const int i = ti * 16 + r, ncols = i + 1;
  float pmax = -3.0e38f;
  for (int j = c; j < ncols; j += 16)
    pmax = fmaxf(pmax, logits[r * LSTRIDE + j]);
  red[r * 16 + c] = pmax;
  __syncthreads();
  if (c == 0) {
    float m = red[r * 16];
#pragma unroll
    for (int q = 1; q < 16; ++q) m = fmaxf(m, red[r * 16 + q]);
    rowmax[r] = m;
  }
  __syncthreads();
  const float rm = rowmax[r];
  float psum = 0.0f;
  for (int j = c; j < ncols; j += 16)
    psum += __expf(logits[r * LSTRIDE + j] - rm);
  red[r * 16 + c] = psum;
  __syncthreads();
  if (c == 0) {
    float s = 0.0f;
#pragma unroll
    for (int q = 0; q < 16; ++q) s += red[r * 16 + q];
    rowsum[r] = s;
  }
  __syncthreads();

  // normalize + write attn row (v4f non-temporal: write-once stream, keep
  // L2 for K/V); masked region is exactly 0.0f (matches reference softmax)
  const float inv = 1.0f / rowsum[r];
  float* arow = attnW + ((size_t)bh * L_SEQ + i) * L_SEQ;
  for (int j0 = c * 4; j0 < L_SEQ; j0 += 64) {
    float* lr = &logits[r * LSTRIDE + j0];
    v4f vv;
    vv[0] = (j0 + 0 < ncols) ? __expf(lr[0] - rm) * inv : 0.0f;
    vv[1] = (j0 + 1 < ncols) ? __expf(lr[1] - rm) * inv : 0.0f;
    vv[2] = (j0 + 2 < ncols) ? __expf(lr[2] - rm) * inv : 0.0f;
    vv[3] = (j0 + 3 < ncols) ? __expf(lr[3] - rm) * inv : 0.0f;
    lr[0] = vv[0]; lr[1] = vv[1]; lr[2] = vv[2]; lr[3] = vv[3];  // reuse for attn@V
    __builtin_nontemporal_store(vv, (v4f*)(arow + j0));
  }
  __syncthreads();

  // ctx(16x64) = attn(16x2048) @ V(2048x64); wave w owns K slice w*256..+255
  // B-fragments are single 32B loads from transposed Vt[dh][l]
  v8f cacc[4] = {{}, {}, {}, {}};
  for (int ks = 0; ks < 8; ++ks) {
    const int kbase = wid * 256 + ks * 32;
    v16bf a;
#pragma unroll
    for (int e = 0; e < 16; ++e)
      a[e] = (__bf16)logits[l15 * LSTRIDE + kbase +
                            e + (hi << 4)];
    const int kb = kbase + (hi << 4);
#pragma unroll
    for (int nt = 0; nt < 4; ++nt) {
      v16bf b = *(const v16bf*)(Vtb + (size_t)(nt * 16 + l15) * L_SEQ + kb);
      cacc[nt] = wmma_bf16(a, b, cacc[nt]);
    }
  }
  float* cp = cpart + wid * 1024;
#pragma unroll
  for (int nt = 0; nt < 4; ++nt)
#pragma unroll
    for (int rr = 0; rr < 8; ++rr)
      cp[(rr + (hi << 3)) * 64 + nt * 16 + l15] = cacc[nt][rr];
  __syncthreads();

  const int b_ = bh >> 3, h_ = bh & 7;
  for (int idx = tid; idx < 1024; idx += 256) {
    float s = 0.0f;
#pragma unroll
    for (int w = 0; w < 8; ++w) s += cpart[w * 1024 + idx];
    const int ip = idx >> 6, n = idx & 63;
    ctx[((size_t)(b_ * L_SEQ + ti * 16 + ip)) * D_MODEL + h_ * D_HEAD + n] =
        (__bf16)s;
  }
}

// ------------------------------------------- final projection: out = ctx@W + b
__global__ __launch_bounds__(256)
void fc_kernel(const __bf16* __restrict__ Xb, const __bf16* __restrict__ Wt,
               const float* __restrict__ bias, float* __restrict__ Y) {
  const int wid = threadIdx.x >> 5, lane = threadIdx.x & 31;
  const int hi = lane >> 4, l15 = lane & 15;
  const int gw = blockIdx.x * 8 + wid;
  const int mt = gw >> 5, nt = gw & 31;
  const int m0 = mt * 16, n0 = nt * 16;

  v8f acc = {};
  const __bf16* xr = Xb + (size_t)(m0 + l15) * D_MODEL;
  const __bf16* wr = Wt + (size_t)(n0 + l15) * D_MODEL;
  for (int k0 = 0; k0 < D_MODEL; k0 += 32) {
    v16bf a = a_frag(xr, k0, hi);
    v16bf b = *(const v16bf*)(wr + k0 + (hi << 4));
    acc = wmma_bf16(a, b, acc);
  }
  const int n = n0 + l15;
  const float bv = bias[n];
#pragma unroll
  for (int r = 0; r < 8; ++r) {
    const int m = m0 + r + (hi << 3);
    Y[(size_t)m * D_MODEL + n] = acc[r] + bv;
  }
}

extern "C" void kernel_launch(void* const* d_in, const int* in_sizes, int n_in,
                              void* d_out, int out_size, void* d_ws, size_t ws_size,
                              hipStream_t stream) {
  (void)in_sizes; (void)n_in; (void)out_size; (void)ws_size;
  const float* q_in = (const float*)d_in[0];
  const float* k_in = (const float*)d_in[1];
  const float* v_in = (const float*)d_in[2];
  // d_in[3] = mask (implicit in kernel)
  const float* Wq  = (const float*)d_in[4];
  const float* bq  = (const float*)d_in[5];
  const float* Wk  = (const float*)d_in[6];
  const float* bk  = (const float*)d_in[7];
  const float* Wv  = (const float*)d_in[8];
  const float* bv  = (const float*)d_in[9];
  const float* Wfc = (const float*)d_in[10];
  const float* bfc = (const float*)d_in[11];
  const float* E   = (const float*)d_in[12];

  // workspace carve (all offsets 256B aligned)
  const size_t ACT = (size_t)N_BATCH * L_SEQ * D_MODEL * sizeof(__bf16);  // 4MB
  const size_t WTB = (size_t)D_MODEL * D_MODEL * sizeof(__bf16);          // 512KB
  char* ws = (char*)d_ws;
  size_t off = 0;
  __bf16* Qh   = (__bf16*)(ws + off); off += ACT;
  __bf16* Kh   = (__bf16*)(ws + off); off += ACT;
  __bf16* Vh   = (__bf16*)(ws + off); off += ACT;
  __bf16* Vt   = (__bf16*)(ws + off); off += ACT;
  __bf16* ctx  = (__bf16*)(ws + off); off += ACT;
  __bf16* Xqb  = (__bf16*)(ws + off); off += ACT;
  __bf16* Xkb  = (__bf16*)(ws + off); off += ACT;
  __bf16* Xvb  = (__bf16*)(ws + off); off += ACT;
  __bf16* Wtq  = (__bf16*)(ws + off); off += WTB;
  __bf16* Wtk  = (__bf16*)(ws + off); off += WTB;
  __bf16* Wtv  = (__bf16*)(ws + off); off += WTB;
  __bf16* Wtfc = (__bf16*)(ws + off); off += WTB;
  __bf16* Eb   = (__bf16*)(ws + off); off += (size_t)L_SEQ * D_HEAD * sizeof(__bf16);

  float* out  = (float*)d_out;
  float* attn = out + (size_t)N_BATCH * L_SEQ * D_MODEL;

  const int act4 = N_BATCH * L_SEQ * D_MODEL / 4;   // 524288
  cvt_kernel<<<(act4 + 255) / 256, 256, 0, stream>>>(q_in, Xqb, act4);
  cvt_kernel<<<(act4 + 255) / 256, 256, 0, stream>>>(k_in, Xkb, act4);
  cvt_kernel<<<(act4 + 255) / 256, 256, 0, stream>>>(v_in, Xvb, act4);
  const int e4 = L_SEQ * D_HEAD / 4;
  cvt_kernel<<<(e4 + 255) / 256, 256, 0, stream>>>(E, Eb, e4);

  dim3 wg(16, 16);
  wtrans_kernel<<<wg, 256, 0, stream>>>(Wq, Wtq);
  wtrans_kernel<<<wg, 256, 0, stream>>>(Wk, Wtk);
  wtrans_kernel<<<wg, 256, 0, stream>>>(Wv, Wtv);
  wtrans_kernel<<<wg, 256, 0, stream>>>(Wfc, Wtfc);

  proj_kernel<<<1024, 256, 0, stream>>>(Xqb, Wtq, bq, Qh);
  proj_kernel<<<1024, 256, 0, stream>>>(Xkb, Wtk, bk, Kh);
  proj_kernel<<<1024, 256, 0, stream>>>(Xvb, Wtv, bv, Vh);

  vtrans_kernel<<<dim3(L_SEQ / 32, D_HEAD / 32, N_BATCH * N_HEADS), 256, 0,
                  stream>>>(Vh, Vt);

  const size_t smem = (size_t)(16 * LSTRIDE + 8 * 512 + 256 + 16 + 16 + 8 * 1024) *
                      sizeof(float);
  attn_kernel<<<N_BATCH * N_HEADS * (L_SEQ / 16), 256, smem, stream>>>(
      Qh, Kh, Vt, Eb, attn, ctx);

  fc_kernel<<<1024, 256, 0, stream>>>(ctx, Wtfc, bfc, out);
}
